// MultiHeadAttention_85555748536542
// MI455X (gfx1250) — compile-verified, matrix operand order checked by reading, not run on hardware
//
#include <hip/hip_runtime.h>

// ---------------------------------------------------------------------------
// MI455X (gfx1250) multi-head attention, bf16 WMMA pipeline, f32 accumulate.
//   B=2, S=2048, D_MODEL=1024, H=16, D_HEAD=64
// Stages: cvt(f32->bf16) -> per-head QKV proj (WMMA, TDM-staged weights) ->
//         flash attention (TDM-staged K/V tiles, WMMA scores + online softmax
//         + WMMA PV) -> output proj (WMMA, TDM-staged weights).
// Shared operands (weights, K/V tiles) are staged once per workgroup into LDS
// by the Tensor Data Mover (double-buffered, s_wait_tensorcnt), removing the
// 8x per-wave redundant global reads.  Per-wave-unique A fragments stay as
// direct global b128 loads.  Workspace use: ~64 MB.
// ---------------------------------------------------------------------------

#define BATCH   2
#define SLEN    2048
#define DMODEL  1024
#define NHEAD   16
#define DHEAD   64
#define BS      (BATCH * SLEN)   // 4096

typedef __attribute__((ext_vector_type(16))) __bf16 v16bf;
typedef __attribute__((ext_vector_type(2)))  __bf16 v2bf;
typedef __attribute__((ext_vector_type(8)))  float  v8f;
typedef __attribute__((ext_vector_type(4)))  unsigned int v4u;
typedef __attribute__((ext_vector_type(8)))  int v8i_;
typedef __attribute__((ext_vector_type(4)))  int v4i_;

// WMMA wrapper: D = A(16x32 bf16) x B(32x16 bf16) + C(f32)
__device__ __forceinline__ v8f wmma_bf16(v16bf a, v16bf b, v8f c) {
  return __builtin_amdgcn_wmma_f32_16x16x32_bf16(
      /*neg_a=*/false, a, /*neg_b=*/false, b,
      /*c_mod=*/(short)0, c, /*reuse_a=*/false, /*reuse_b=*/false);
}

// Load a 16x32 bf16 fragment (A layout; also B when columns are stored as
// contiguous rows of an [N][K] matrix).  CDNA5 16-bit A layout: lane holds
// row = lane&15; VGPR j holds K = 2j + (j>=4?8:0) + (lane>=16?8:0).
__device__ __forceinline__ v16bf load_frag(const __bf16* __restrict__ p, int ld) {
  const int lane = threadIdx.x & 31;
  const int row  = lane & 15;
  const int koff = (lane & 16) ? 8 : 0;
  const __bf16* r = p + (size_t)row * ld;
  v16bf a;
#pragma unroll
  for (int j = 0; j < 8; ++j) {
    const int kk = 2 * j + ((j >= 4) ? 8 : 0) + koff;
    v2bf t = *(const v2bf*)(r + kk);
    a[2 * j]     = t[0];
    a[2 * j + 1] = t[1];
  }
  return a;
}

// ---------------------------------------------------------------------------
// Tensor Data Mover: async 2D tile load global -> LDS.
// D# per CDNA5 ISA ch.8: group0 = {count=1, lds_addr, global_addr, type=2},
// group1 = {data_size=2B, tensor dims = tile dims (tile fully interior),
// tile_dim0/1, tensor_dim0_stride}.  Issued by one wave (EXEC ignored by TDM).
__device__ __forceinline__ void tdm_load_2d(const __bf16* gptr, __bf16* lptr,
                                            unsigned tile_d0, unsigned tile_d1,
                                            unsigned stride_elems) {
  unsigned lds = (unsigned)(size_t)lptr;  // generic shared ptr: low 32 = LDS byte offset
  unsigned long long ga = (unsigned long long)(size_t)gptr;
  v4u g0;
  g0[0] = 1u;                                         // count=1 (valid user D#)
  g0[1] = lds;                                        // lds_addr [63:32]
  g0[2] = (unsigned)(ga & 0xFFFFFFFFu);               // global_addr [95:64]
  g0[3] = (unsigned)((ga >> 32) & 0x01FFFFFFu) | 0x80000000u;  // addr hi | type=2
  v8i_ g1;
  g1[0] = (int)(1u << 16);                            // wg_mask=0, data_size=1 (2B)
  g1[1] = (int)((tile_d0 & 0xFFFFu) << 16);           // tensor_dim0[15:0] @bit48
  g1[2] = (int)(((tile_d0 >> 16) & 0xFFFFu) | ((tile_d1 & 0xFFFFu) << 16)); // dim0 hi, dim1 lo
  g1[3] = (int)(((tile_d1 >> 16) & 0xFFFFu) | ((tile_d0 & 0xFFFFu) << 16)); // dim1 hi, tile_dim0
  g1[4] = (int)(tile_d1 & 0xFFFFu);                   // tile_dim1; tile_dim2=0
  g1[5] = (int)stride_elems;                          // tensor_dim0_stride lo32
  g1[6] = 0;                                          // stride hi16, dim1_stride lo
  g1[7] = 0;
  v4i_ z4 = {0, 0, 0, 0};
#if __clang_major__ >= 23
  v8i_ z8 = {0, 0, 0, 0, 0, 0, 0, 0};
  __builtin_amdgcn_tensor_load_to_lds(g0, g1, z4, z4, z8, 0);
#else
  __builtin_amdgcn_tensor_load_to_lds(g0, g1, z4, z4, 0);
#endif
}

// ---------------------------------------------------------------------------
// Elementwise f32 -> bf16
__global__ void cvt_kernel(const float* __restrict__ src, __bf16* __restrict__ dst, int n) {
  int i = blockIdx.x * 256 + threadIdx.x;
  if (i < n) dst[i] = (__bf16)src[i];
}

// f32 [H][K][N] -> bf16 transposed [H][N][K]
__global__ void cvt_tr_kernel(const float* __restrict__ W, __bf16* __restrict__ Wt,
                              int H, int K, int N) {
  long idx = (long)blockIdx.x * 256 + threadIdx.x;
  long total = (long)H * K * N;
  if (idx >= total) return;
  int  n  = (int)(idx % N);
  long t  = idx / N;
  int  kk = (int)(t % K);
  int  h  = (int)(t / K);
  Wt[((long)h * N + n) * K + kk] = (__bf16)W[idx];
}

// ---------------------------------------------------------------------------
// Per-head projection: Y[b,h,s,e] = sum_d X[b,s,d] * W[h,d,e] + bias[h,e]
// X: bf16 [BS][DMODEL]; Wt: bf16 [H][DHEAD][DMODEL]; bias: f32 [H][DHEAD].
// Weight K-tiles (64 rows x 32 K, 4 KB) are TDM-staged into LDS, double-buffered.
// TRANS=0 -> Y bf16 [B,H,S,E]; TRANS=1 -> Y bf16 [B,H,E,S] (for V).
template <int TRANS>
__global__ __launch_bounds__(256) void proj_kernel(const __bf16* __restrict__ X,
                                                   const __bf16* __restrict__ Wt,
                                                   const float*  __restrict__ bias,
                                                   __bf16* __restrict__ Y) {
  __shared__ __align__(128) __bf16 wl[2][DHEAD * 32];
  const int lane = threadIdx.x & 31;
  const int wave = threadIdx.x >> 5;
  const int h    = blockIdx.y;
  const int mrow = blockIdx.x * 128 + wave * 16;

  v8f acc[4];
#pragma unroll
  for (int j = 0; j < 4; ++j) acc[j] = {};

  const __bf16* Wh = Wt + (size_t)h * DHEAD * DMODEL;
  const int NT = DMODEL / 32;

  if (wave == 0) tdm_load_2d(Wh, &wl[0][0], 32, DHEAD, DMODEL);

  for (int i = 0; i < NT; ++i) {
    const int kk = i * 32;
    if (wave == 0) {
      if (i + 1 < NT) {
        tdm_load_2d(Wh + (i + 1) * 32, &wl[(i + 1) & 1][0], 32, DHEAD, DMODEL);
        __builtin_amdgcn_s_wait_tensorcnt((short)1);   // tile i complete
      } else {
        __builtin_amdgcn_s_wait_tensorcnt((short)0);
      }
    }
    __syncthreads();

    v16bf a = load_frag(X + (size_t)mrow * DMODEL + kk, DMODEL);
    const __bf16* wb = &wl[i & 1][0];
#pragma unroll
    for (int j = 0; j < 4; ++j) {
      v16bf b = load_frag(wb + (j * 16) * 32, 32);
      acc[j] = wmma_bf16(a, b, acc[j]);
    }
    __syncthreads();   // all waves done with buf before it is refilled
  }

  const int hi  = (lane & 16) ? 8 : 0;
  const int col = lane & 15;
#pragma unroll
  for (int j = 0; j < 4; ++j) {
    const int e = j * 16 + col;
    const float bv = bias[h * DHEAD + e];
#pragma unroll
    for (int r = 0; r < 8; ++r) {
      const int gm = mrow + r + hi;          // flat row in [BS]
      const int b  = gm >> 11;               // / SLEN
      const int s  = gm & (SLEN - 1);
      const float val = acc[j][r] + bv;
      if (TRANS)
        Y[(((size_t)(b * NHEAD + h)) * DHEAD + e) * SLEN + s] = (__bf16)val;
      else
        Y[(((size_t)(b * NHEAD + h)) * SLEN + s) * DHEAD + e] = (__bf16)val;
    }
  }
}

// ---------------------------------------------------------------------------
// Flash attention per (b,h): 8 waves x 16 queries = 128 queries per block.
// K and V 64x64 tiles are TDM-staged into LDS (double-buffered) and shared by
// all 8 waves.  qh,kh: bf16 [B,H,S,E]; vt: bf16 [B,H,E,S]; out bf16 [BS][DMODEL].
__global__ __launch_bounds__(256) void attn_kernel(const __bf16* __restrict__ qh,
                                                   const __bf16* __restrict__ kh,
                                                   const __bf16* __restrict__ vt,
                                                   __bf16* __restrict__ attnb) {
  __shared__ __align__(128) __bf16 klds[2][64 * 64];   // [key][e]
  __shared__ __align__(128) __bf16 vlds[2][64 * 64];   // [e][key]
  __shared__ __align__(128) __bf16 plds[8 * 16 * 64];  // wave-private P tiles
  const int lane = threadIdx.x & 31;
  const int wave = threadIdx.x >> 5;
  const int h  = blockIdx.y;
  const int b  = blockIdx.z;
  const int q0 = blockIdx.x * 128 + wave * 16;
  const int bh = b * NHEAD + h;

  const __bf16* qbh = qh + (size_t)bh * SLEN * DHEAD;
  const __bf16* kbh = kh + (size_t)bh * SLEN * DHEAD;
  const __bf16* vbh = vt + (size_t)bh * DHEAD * SLEN;
  __bf16* pw = plds + wave * (16 * 64);

  // Q fragments (K = 64 -> two 16x32 fragments), loaded once.
  const v16bf aq0 = load_frag(qbh + (size_t)q0 * DHEAD, DHEAD);
  const v16bf aq1 = load_frag(qbh + (size_t)q0 * DHEAD + 32, DHEAD);

  v8f oacc[4];
  float m[8], l[8];
#pragma unroll
  for (int j = 0; j < 4; ++j) oacc[j] = {};
#pragma unroll
  for (int r = 0; r < 8; ++r) { m[r] = -1e30f; l[r] = 0.f; }

  const int hi  = (lane & 16) ? 8 : 0;
  const int col = lane & 15;
  const int NT  = SLEN / 64;

  if (wave == 0) {
    tdm_load_2d(kbh, &klds[0][0], 64, 64, DHEAD);       // keys 0..63, contiguous rows
    tdm_load_2d(vbh, &vlds[0][0], 64, 64, SLEN);        // e rows, key cols 0..63
  }

  for (int i = 0; i < NT; ++i) {
    const int t0 = i * 64;
    if (wave == 0) {
      if (i + 1 < NT) {
        tdm_load_2d(kbh + (size_t)(t0 + 64) * DHEAD, &klds[(i + 1) & 1][0], 64, 64, DHEAD);
        tdm_load_2d(vbh + (t0 + 64),                 &vlds[(i + 1) & 1][0], 64, 64, SLEN);
        __builtin_amdgcn_s_wait_tensorcnt((short)2);    // tile i's two loads done
      } else {
        __builtin_amdgcn_s_wait_tensorcnt((short)0);
      }
    }
    __syncthreads();
    const __bf16* kb_l = &klds[i & 1][0];
    const __bf16* vb_l = &vlds[i & 1][0];

    // scores: S[16 x 64] = Q(16x64) . K^T   (B cols = keys, rows of kb_l)
    v8f sacc[4];
#pragma unroll
    for (int j = 0; j < 4; ++j) sacc[j] = {};
#pragma unroll
    for (int j = 0; j < 4; ++j) {
      v16bf b0 = load_frag(kb_l + (j * 16) * 64, 64);
      sacc[j] = wmma_bf16(aq0, b0, sacc[j]);
      v16bf b1 = load_frag(kb_l + (j * 16) * 64 + 32, 64);
      sacc[j] = wmma_bf16(aq1, b1, sacc[j]);
    }

    // online softmax (row lives across 16 lanes of a half; VGPR r = row)
    float p[4][8];
#pragma unroll
    for (int r = 0; r < 8; ++r) {
      float mx = -1e30f;
#pragma unroll
      for (int j = 0; j < 4; ++j) mx = fmaxf(mx, sacc[j][r]);
      mx *= 0.125f;                                    // 1/sqrt(64)
#pragma unroll
      for (int st = 1; st < 16; st <<= 1) mx = fmaxf(mx, __shfl_xor(mx, st, 32));
      const float mnew  = fmaxf(m[r], mx);
      const float alpha = __expf(m[r] - mnew);
      float rs = 0.f;
#pragma unroll
      for (int j = 0; j < 4; ++j) {
        p[j][r] = __expf(sacc[j][r] * 0.125f - mnew);
        rs += p[j][r];
      }
#pragma unroll
      for (int st = 1; st < 16; st <<= 1) rs += __shfl_xor(rs, st, 32);
      l[r] = alpha * l[r] + rs;
      m[r] = mnew;
#pragma unroll
      for (int j = 0; j < 4; ++j) oacc[j][r] *= alpha;
    }

    // C-layout -> A-layout via wave-private LDS tile (bf16)
#pragma unroll
    for (int j = 0; j < 4; ++j)
#pragma unroll
      for (int r = 0; r < 8; ++r)
        pw[(r + hi) * 64 + j * 16 + col] = (__bf16)p[j][r];
    asm volatile("s_wait_dscnt 0" ::: "memory");

    const v16bf ap0 = load_frag(pw, 64);
    const v16bf ap1 = load_frag(pw + 32, 64);

    // O += P(16x64) . V(64xE), V columns = rows of vb_l
#pragma unroll
    for (int j = 0; j < 4; ++j) {
      v16bf b0 = load_frag(vb_l + (j * 16) * 64, 64);
      oacc[j] = wmma_bf16(ap0, b0, oacc[j]);
      v16bf b1 = load_frag(vb_l + (j * 16) * 64 + 32, 64);
      oacc[j] = wmma_bf16(ap1, b1, oacc[j]);
    }
    __syncthreads();   // all waves done with K/V buf before refill
  }

  // normalize and write concat-head buffer [BS][DMODEL]
#pragma unroll
  for (int j = 0; j < 4; ++j)
#pragma unroll
    for (int r = 0; r < 8; ++r) {
      const int srow = q0 + r + hi;
      const float val = oacc[j][r] / l[r];
      attnb[((size_t)(b * SLEN + srow)) * DMODEL + h * DHEAD + j * 16 + col] = (__bf16)val;
    }
}

// ---------------------------------------------------------------------------
// Output projection: out[m,n] = sum_d attn[m,d] * Wo[d,n] + bo[n]   (f32 out)
// Weight K-tiles TDM-staged into LDS, double-buffered, shared by all waves.
__global__ __launch_bounds__(256) void outproj_kernel(const __bf16* __restrict__ A,
                                                      const __bf16* __restrict__ Wot,
                                                      const float*  __restrict__ bo,
                                                      float* __restrict__ out) {
  __shared__ __align__(128) __bf16 wl[2][64 * 32];
  const int lane = threadIdx.x & 31;
  const int wave = threadIdx.x >> 5;
  const int mrow = blockIdx.x * 128 + wave * 16;
  const int n0   = blockIdx.y * 64;

  v8f acc[4];
#pragma unroll
  for (int j = 0; j < 4; ++j) acc[j] = {};

  const __bf16* Wn = Wot + (size_t)n0 * DMODEL;
  const int NT = DMODEL / 32;

  if (wave == 0) tdm_load_2d(Wn, &wl[0][0], 32, 64, DMODEL);

  for (int i = 0; i < NT; ++i) {
    const int kk = i * 32;
    if (wave == 0) {
      if (i + 1 < NT) {
        tdm_load_2d(Wn + (i + 1) * 32, &wl[(i + 1) & 1][0], 32, 64, DMODEL);
        __builtin_amdgcn_s_wait_tensorcnt((short)1);
      } else {
        __builtin_amdgcn_s_wait_tensorcnt((short)0);
      }
    }
    __syncthreads();

    v16bf a = load_frag(A + (size_t)mrow * DMODEL + kk, DMODEL);
    const __bf16* wb = &wl[i & 1][0];
#pragma unroll
    for (int j = 0; j < 4; ++j) {
      v16bf b = load_frag(wb + (j * 16) * 32, 32);
      acc[j] = wmma_bf16(a, b, acc[j]);
    }
    __syncthreads();
  }

  const int hi  = (lane & 16) ? 8 : 0;
  const int col = lane & 15;
#pragma unroll
  for (int j = 0; j < 4; ++j) {
    const int n = n0 + j * 16 + col;
    const float bias = bo[n];
#pragma unroll
    for (int r = 0; r < 8; ++r) {
      const int gm = mrow + r + hi;
      out[(size_t)gm * DMODEL + n] = acc[j][r] + bias;
    }
  }
}

// ---------------------------------------------------------------------------
extern "C" void kernel_launch(void* const* d_in, const int* in_sizes, int n_in,
                              void* d_out, int out_size, void* d_ws, size_t ws_size,
                              hipStream_t stream) {
  const float* q  = (const float*)d_in[0];
  const float* k  = (const float*)d_in[1];
  const float* v  = (const float*)d_in[2];
  const float* Wq = (const float*)d_in[3];
  const float* bq = (const float*)d_in[4];
  const float* Wk = (const float*)d_in[5];
  const float* bk = (const float*)d_in[6];
  const float* Wv = (const float*)d_in[7];
  const float* bv = (const float*)d_in[8];
  const float* Wo = (const float*)d_in[9];
  const float* bo = (const float*)d_in[10];
  float* out = (float*)d_out;

  // Workspace carve-up (bf16 elements); total = 64 MB.
  char* ws = (char*)d_ws;
  size_t off = 0;
  auto take = [&](size_t elems) {
    __bf16* p = (__bf16*)(ws + off);
    off += elems * sizeof(__bf16);
    return p;
  };
  __bf16* qb    = take((size_t)BS * DMODEL);
  __bf16* kb    = take((size_t)BS * DMODEL);
  __bf16* vb    = take((size_t)BS * DMODEL);
  __bf16* Wqt   = take((size_t)NHEAD * DHEAD * DMODEL);
  __bf16* Wkt   = take((size_t)NHEAD * DHEAD * DMODEL);
  __bf16* Wvt   = take((size_t)NHEAD * DHEAD * DMODEL);
  __bf16* Wot   = take((size_t)DMODEL * DMODEL);
  __bf16* qhp   = take((size_t)BATCH * NHEAD * SLEN * DHEAD);
  __bf16* khp   = take((size_t)BATCH * NHEAD * SLEN * DHEAD);
  __bf16* vtp   = take((size_t)BATCH * NHEAD * DHEAD * SLEN);
  __bf16* attnb = take((size_t)BS * DMODEL);
  (void)ws_size; (void)in_sizes; (void)n_in; (void)out_size;

  const int nx = BS * DMODEL;  // 4,194,304
  cvt_kernel<<<(nx + 255) / 256, 256, 0, stream>>>(q, qb, nx);
  cvt_kernel<<<(nx + 255) / 256, 256, 0, stream>>>(k, kb, nx);
  cvt_kernel<<<(nx + 255) / 256, 256, 0, stream>>>(v, vb, nx);

  const int nw = NHEAD * DMODEL * DHEAD;  // 1,048,576
  cvt_tr_kernel<<<(nw + 255) / 256, 256, 0, stream>>>(Wq, Wqt, NHEAD, DMODEL, DHEAD);
  cvt_tr_kernel<<<(nw + 255) / 256, 256, 0, stream>>>(Wk, Wkt, NHEAD, DMODEL, DHEAD);
  cvt_tr_kernel<<<(nw + 255) / 256, 256, 0, stream>>>(Wv, Wvt, NHEAD, DMODEL, DHEAD);
  const int no = DMODEL * DMODEL;
  cvt_tr_kernel<<<(no + 255) / 256, 256, 0, stream>>>(Wo, Wot, 1, DMODEL, DMODEL);

  dim3 pgrid(BS / 128, NHEAD);
  proj_kernel<0><<<pgrid, 256, 0, stream>>>(qb, Wqt, bq, qhp);
  proj_kernel<0><<<pgrid, 256, 0, stream>>>(kb, Wkt, bk, khp);
  proj_kernel<1><<<pgrid, 256, 0, stream>>>(vb, Wvt, bv, vtp);

  attn_kernel<<<dim3(SLEN / 128, NHEAD, BATCH), 256, 0, stream>>>(qhp, khp, vtp, attnb);

  outproj_kernel<<<dim3(BS / 128, DMODEL / 64), 256, 0, stream>>>(attnb, Wot, bo, out);
}
